// Sequence_33474975105039
// MI455X (gfx1250) — matrix operand order repaired and checked by reading, not compiled, not run on hardware
//
#include <hip/hip_runtime.h>

typedef __attribute__((ext_vector_type(16))) _Float16 v16h;
typedef __attribute__((ext_vector_type(8)))  _Float16 v8h;
typedef __attribute__((ext_vector_type(8)))  float    v8f;

#define HID 51
#define NT  13          // N tiles: 208/16 (204 padded)
#define WAVES 4
#define THREADS 128
#define ROWS_PER_BLOCK 64   // WAVES * 16

#define WAVE_LDS_FENCE() asm volatile("s_wait_dscnt 0" ::: "memory")

__device__ __forceinline__ v8f wmma16(v16h a, v16h b, v8f c) {
  return __builtin_amdgcn_wmma_f32_16x16x32_f16(false, a, false, b, (short)0, c,
                                                false, false);
}
__device__ __forceinline__ float fsig(float x) { return 1.0f / (1.0f + __expf(-x)); }
__device__ __forceinline__ float ftanh(float x) {
  float e = __expf(2.0f * x);
  return (e - 1.0f) / (e + 1.0f);
}

// Load one 16x32 f16 A-fragment for this lane from row-major LDS staging.
// A layout (matches B packing): lane<16 holds K = 0..15 +32*kt of row M=lane;
// lane>=16 holds K = 16..31 +32*kt of row M=lane-16.
__device__ __forceinline__ v16h loadA(const _Float16* rowBase, int kt, int bo) {
  union { v16h v; v8h h[2]; } u;
  u.h[0] = *(const v8h*)(rowBase + kt * 32 + 2 * bo);
  u.h[1] = *(const v8h*)(rowBase + kt * 32 + 2 * bo + 8);
  return u.v;
}

__global__ __launch_bounds__(THREADS) void lstm_seq_kernel(
    const float* __restrict__ input, const float* __restrict__ h1_0,
    const float* __restrict__ c1_0, const float* __restrict__ h2_0,
    const float* __restrict__ c2_0, const float* __restrict__ Wih1,
    const float* __restrict__ Whh1, const float* __restrict__ bih1,
    const float* __restrict__ bhh1, const float* __restrict__ Wih2,
    const float* __restrict__ Whh2, const float* __restrict__ bih2,
    const float* __restrict__ bhh2, const float* __restrict__ Wl,
    const float* __restrict__ blp, float* __restrict__ out,
    int Bsz, int Tin, int future)
{
  // Weights pre-packed into WMMA B-fragment order: [frag][lane][16 halfs].
  __shared__ __align__(32) _Float16 sB1[26][32][16];   // 64x208 padded B1
  __shared__ __align__(32) _Float16 sB2[52][32][16];   // 128x208 padded B2
  __shared__ float sBias1[208], sBias2[208], sWl[52];
  // Per-wave A staging (row-major f16) and fp32 state.
  __shared__ __align__(16) _Float16 sA1[WAVES][16][64];
  __shared__ __align__(16) _Float16 sA2[WAVES][16][128];
  __shared__ float sGates[WAVES][16][208];
  __shared__ float sC1[WAVES][16][HID], sC2[WAVES][16][HID];
  __shared__ float sH1[WAVES][16][HID], sH2[WAVES][16][HID];

  const int tid  = threadIdx.x;
  const int lane = tid & 31;
  const int wave = tid >> 5;
  const long rowBase = (long)blockIdx.x * ROWS_PER_BLOCK;
  const float blv = blp[0];

  // ---------------- one-time init: pack weights / load state ----------------
  // B-fragment layout (contiguous k, as in loadA): lane<16 holds K=0..15 (elt e -> K=e),
  // lane>=16 holds K=16..31, N = lane&15, all + 32*kt / 16*nt offsets.
  for (int idx = tid; idx < 26 * 32; idx += THREADS) {
    int frag = idx >> 5, l = idx & 31;
    int nt = frag >> 1, kt = frag & 1;
    int N = nt * 16 + (l & 15);
    int Kb = (l >= 16) ? 16 : 0;
#pragma unroll
    for (int e = 0; e < 16; ++e) {
      int K = kt * 32 + Kb + e;
      float v = 0.0f;
      if (N < 204) {
        if (K == 0)       v = Wih1[N];                 // Wih1 is (204,1)
        else if (K <= HID) v = Whh1[N * HID + (K - 1)];
      }
      sB1[frag][l][e] = (_Float16)v;
    }
  }
  for (int idx = tid; idx < 52 * 32; idx += THREADS) {
    int frag = idx >> 5, l = idx & 31;
    int nt = frag >> 2, kt = frag & 3;
    int N = nt * 16 + (l & 15);
    int Kb = (l >= 16) ? 16 : 0;
#pragma unroll
    for (int e = 0; e < 16; ++e) {
      int K = kt * 32 + Kb + e;
      float v = 0.0f;
      if (N < 204) {
        if (K < HID)                     v = Wih2[N * HID + K];
        else if (K >= 64 && K < 64 + HID) v = Whh2[N * HID + (K - 64)];
      }
      sB2[frag][l][e] = (_Float16)v;
    }
  }
  for (int n = tid; n < 208; n += THREADS) {
    sBias1[n] = (n < 204) ? (bih1[n] + bhh1[n]) : 0.0f;
    sBias2[n] = (n < 204) ? (bih2[n] + bhh2[n]) : 0.0f;
  }
  for (int u = tid; u < 52; u += THREADS) sWl[u] = (u < HID) ? Wl[u] : 0.0f;
  for (int i = tid; i < WAVES * 16 * 64;  i += THREADS) (&sA1[0][0][0])[i] = (_Float16)0.0f;
  for (int i = tid; i < WAVES * 16 * 128; i += THREADS) (&sA2[0][0][0])[i] = (_Float16)0.0f;
  __syncthreads();
  for (int p = tid; p < ROWS_PER_BLOCK * HID; p += THREADS) {
    int ml = p / HID, u = p - ml * HID;
    int w = ml >> 4, mw = ml & 15;
    long r = (rowBase + ml) * HID + u;
    float h1 = h1_0[r], c1 = c1_0[r], h2 = h2_0[r], c2 = c2_0[r];
    sC1[w][mw][u] = c1;  sC2[w][mw][u] = c2;
    sH1[w][mw][u] = h1;  sH2[w][mw][u] = h2;
    sA1[w][mw][1 + u]  = (_Float16)h1;   // layer-1 recurrent input
    sA2[w][mw][64 + u] = (_Float16)h2;   // layer-2 recurrent input
  }
  __syncthreads();

  // ---------------- recurrent loop: 1 + future steps ----------------
  const _Float16* a1Row = &sA1[wave][lane & 15][0];
  const _Float16* a2Row = &sA2[wave][lane & 15][0];
  const int bo = (lane >= 16) ? 8 : 0;           // A-fragment half offset
  const int Mb = (lane >= 16) ? 8 : 0;           // C/D fragment row offset
  const int Nl = lane & 15;
  float xv = 0.0f;

#pragma unroll 1
  for (int t = 0; t <= future; ++t) {
    // feed x (conditioning input at t==0, else previous prediction)
    if (lane < 16) {
      if (t == 0) xv = input[(long)(Tin - 1) * Bsz + rowBase + wave * 16 + lane];
      sA1[wave][lane][0] = (_Float16)xv;
    }
    WAVE_LDS_FENCE();

    // ---- layer 1: gates = A1(16x64) x B1(64x208) ----
    {
      v16h a0 = loadA(a1Row, 0, bo);
      v16h a1 = loadA(a1Row, 1, bo);
      float* g = &sGates[wave][0][0];
#pragma unroll
      for (int nt = 0; nt < NT; ++nt) {
        v8f acc = {0.f, 0.f, 0.f, 0.f, 0.f, 0.f, 0.f, 0.f};
        acc = wmma16(a0, *(const v16h*)&sB1[nt * 2 + 0][lane][0], acc);
        acc = wmma16(a1, *(const v16h*)&sB1[nt * 2 + 1][lane][0], acc);
        int N = nt * 16 + Nl;
#pragma unroll
        for (int r = 0; r < 8; ++r) g[(Mb + r) * 208 + N] = acc[r];
      }
    }
    WAVE_LDS_FENCE();

    // ---- layer 1 cell elementwise ----
    for (int p = lane; p < 16 * HID; p += 32) {
      int m = p / HID, u = p - m * HID;
      float gi = sGates[wave][m][u]        + sBias1[u];
      float gf = sGates[wave][m][51 + u]   + sBias1[51 + u];
      float gg = sGates[wave][m][102 + u]  + sBias1[102 + u];
      float go = sGates[wave][m][153 + u]  + sBias1[153 + u];
      float c  = fsig(gf) * sC1[wave][m][u] + fsig(gi) * ftanh(gg);
      float h  = fsig(go) * ftanh(c);
      sC1[wave][m][u] = c;
      sH1[wave][m][u] = h;
      sA1[wave][m][1 + u] = (_Float16)h;   // next-step layer-1 recurrence
      sA2[wave][m][u]     = (_Float16)h;   // layer-2 input this step
    }
    WAVE_LDS_FENCE();

    // ---- layer 2: gates = A2(16x128) x B2(128x208) ----
    {
      v16h a0 = loadA(a2Row, 0, bo);
      v16h a1 = loadA(a2Row, 1, bo);
      v16h a2 = loadA(a2Row, 2, bo);
      v16h a3 = loadA(a2Row, 3, bo);
      float* g = &sGates[wave][0][0];
#pragma unroll
      for (int nt = 0; nt < NT; ++nt) {
        v8f acc = {0.f, 0.f, 0.f, 0.f, 0.f, 0.f, 0.f, 0.f};
        acc = wmma16(a0, *(const v16h*)&sB2[nt * 4 + 0][lane][0], acc);
        acc = wmma16(a1, *(const v16h*)&sB2[nt * 4 + 1][lane][0], acc);
        acc = wmma16(a2, *(const v16h*)&sB2[nt * 4 + 2][lane][0], acc);
        acc = wmma16(a3, *(const v16h*)&sB2[nt * 4 + 3][lane][0], acc);
        int N = nt * 16 + Nl;
#pragma unroll
        for (int r = 0; r < 8; ++r) g[(Mb + r) * 208 + N] = acc[r];
      }
    }
    WAVE_LDS_FENCE();

    // ---- layer 2 cell elementwise ----
    for (int p = lane; p < 16 * HID; p += 32) {
      int m = p / HID, u = p - m * HID;
      float gi = sGates[wave][m][u]        + sBias2[u];
      float gf = sGates[wave][m][51 + u]   + sBias2[51 + u];
      float gg = sGates[wave][m][102 + u]  + sBias2[102 + u];
      float go = sGates[wave][m][153 + u]  + sBias2[153 + u];
      float c  = fsig(gf) * sC2[wave][m][u] + fsig(gi) * ftanh(gg);
      float h  = fsig(go) * ftanh(c);
      sC2[wave][m][u] = c;
      sH2[wave][m][u] = h;
      sA2[wave][m][64 + u] = (_Float16)h;  // next-step layer-2 recurrence
    }
    WAVE_LDS_FENCE();

    // ---- output projection: out = h2 . Wl + bl (16 rows / wave) ----
    if (lane < 16) {
      float s = blv;
#pragma unroll 1
      for (int u = 0; u < HID; ++u) s += sWl[u] * sH2[wave][lane][u];
      out[(rowBase + wave * 16 + lane) * (long)(future + 1) + t] = s;
      xv = s;
    }
  }

  // ---------------- final states ----------------
  __syncthreads();
  const size_t S  = (size_t)Bsz * (size_t)(future + 1);
  const size_t BH = (size_t)Bsz * HID;
  for (int p = tid; p < ROWS_PER_BLOCK * HID; p += THREADS) {
    int ml = p / HID, u = p - ml * HID;
    int w = ml >> 4, mw = ml & 15;
    size_t r = (size_t)(rowBase + ml) * HID + u;
    out[S + r]          = sH1[w][mw][u];
    out[S + BH + r]     = sC1[w][mw][u];
    out[S + 2 * BH + r] = sH2[w][mw][u];
    out[S + 3 * BH + r] = sC2[w][mw][u];
  }
}

extern "C" void kernel_launch(void* const* d_in, const int* in_sizes, int n_in,
                              void* d_out, int out_size, void* d_ws, size_t ws_size,
                              hipStream_t stream) {
  (void)n_in; (void)d_ws; (void)ws_size;
  const int Bsz = in_sizes[1] / HID;                       // 4096
  const int Tin = in_sizes[0] / Bsz;                       // 32
  const int future =
      (int)(((size_t)out_size - 4 * (size_t)Bsz * HID) / Bsz) - 1;  // 128
  dim3 grid(Bsz / ROWS_PER_BLOCK), block(THREADS);
  hipLaunchKernelGGL(lstm_seq_kernel, grid, block, 0, stream,
                     (const float*)d_in[0], (const float*)d_in[1],
                     (const float*)d_in[2], (const float*)d_in[3],
                     (const float*)d_in[4], (const float*)d_in[5],
                     (const float*)d_in[6], (const float*)d_in[7],
                     (const float*)d_in[8], (const float*)d_in[9],
                     (const float*)d_in[10], (const float*)d_in[11],
                     (const float*)d_in[12], (const float*)d_in[13],
                     (const float*)d_in[14], (float*)d_out,
                     Bsz, Tin, future);
}
